// MessagePassingConvolution_84859963834518
// MI455X (gfx1250) — compile-verified, matrix operand order checked by reading, not run on hardware
//
#include <hip/hip_runtime.h>
#include <stdint.h>

// ---------------- problem constants (from reference) ----------------
#define MULF      64
#define NBASIS    8
#define HID       64
#define NIRR      192
#define OUTF      576          // 64 + 64*3 + 64*5
#define EDGES_PER_BLOCK 128    // 8 waves * 16 edges

#define SILU_NORM 1.6791f      // 1/sqrt(E[silu(x)^2]), x~N(0,1)
#define SQRT3     1.7320508f
#define C15       3.8729833f   // sqrt(15)
#define C5H       1.1180340f   // sqrt(5)/2
#define C15H      1.9364917f   // sqrt(15)/2

typedef __attribute__((ext_vector_type(16))) __bf16        v16bf;
typedef __attribute__((ext_vector_type(8)))  float         v8f;
typedef __attribute__((ext_vector_type(4)))  unsigned int  u32x4;

union Frag {
    v16bf          v;
    u32x4          q[2];
    unsigned short s[16];
};

__device__ __forceinline__ unsigned short f2bf(float x) {
    unsigned int u = __builtin_bit_cast(unsigned int, x);
    unsigned int r = u + 0x7FFFu + ((u >> 16) & 1u);   // round-to-nearest-even
    return (unsigned short)(r >> 16);
}

// fast silu * SILU_NORM: x * sigmoid(x), sigmoid via v_exp_f32 + v_rcp_f32
__device__ __forceinline__ float silu_norm(float x) {
    float s = __builtin_amdgcn_rcpf(1.0f + __expf(-x));
    return SILU_NORM * x * s;
}

__device__ __forceinline__ Frag lds_frag(const unsigned short* p) {
    Frag f;
    f.q[0] = *(const u32x4*)(p);        // K = base .. base+7
    f.q[1] = *(const u32x4*)(p + 16);   // K = base+16 .. base+23
    return f;
}

__device__ __forceinline__ v8f wmma_bf16(const Frag& a, const Frag& b, v8f c) {
    return __builtin_amdgcn_wmma_f32_16x16x32_bf16(
        /*neg_a=*/false, a.v, /*neg_b=*/false, b.v,
        /*c_mod=*/(short)0, c, /*reuse_a=*/false, /*reuse_b=*/false);
}

__device__ __forceinline__ void atomic_add_f32(float* p, float v) {
    (void)__hip_atomic_fetch_add(p, v, __ATOMIC_RELAXED, __HIP_MEMORY_SCOPE_AGENT);
}

// one hidden layer: act(16x64 bf16 in LDS) @ Wt(64x64 bf16, [N][K] in LDS) -> silu -> act
__device__ __forceinline__ void hidden_layer(const unsigned short* __restrict__ Wt,
                                             unsigned short* __restrict__ act,
                                             int l15, int kb, int half) {
    Frag af[2];
#pragma unroll
    for (int ks = 0; ks < 2; ++ks)
        af[ks] = lds_frag(&act[l15 * HID + ks * 32 + kb]);

    v8f acc[4];
#pragma unroll
    for (int nb = 0; nb < 4; ++nb) {
        v8f c = {};
#pragma unroll
        for (int ks = 0; ks < 2; ++ks)
            c = wmma_bf16(af[ks], lds_frag(&Wt[(nb * 16 + l15) * HID + ks * 32 + kb]), c);
        acc[nb] = c;
    }
    __syncthreads();
#pragma unroll
    for (int nb = 0; nb < 4; ++nb)
#pragma unroll
        for (int r = 0; r < 8; ++r)
            act[(half * 8 + r) * HID + nb * 16 + l15] = f2bf(silu_norm(acc[nb][r]));
    __syncthreads();
}

__global__ __launch_bounds__(256)
void mace_msgpass_kernel(const float* __restrict__ vectors,
                         const float* __restrict__ node_feats,
                         const float* __restrict__ radial,
                         const float* __restrict__ W0,
                         const float* __restrict__ W1,
                         const float* __restrict__ W2,
                         const float* __restrict__ W3,
                         const int*   __restrict__ senders,
                         const int*   __restrict__ receivers,
                         float*       __restrict__ out) {
    // LDS: weights (bf16, transposed [N][K], prescaled) + per-wave activation tiles
    __shared__ unsigned short sW0t[HID * 32];        //  4 KB (K padded 8->32)
    __shared__ unsigned short sW1t[HID * HID];       //  8 KB
    __shared__ unsigned short sW2t[HID * HID];       //  8 KB
    __shared__ unsigned short sW3t[NIRR * HID];      // 24 KB
    __shared__ unsigned short sAct[8 * 16 * HID];    // 16 KB

    const int tid = threadIdx.x;

    // ---- stage weights: transpose + prescale + bf16 ----
    for (int i = tid; i < HID * 32; i += 256) {
        int n = i >> 5, k = i & 31;
        float v = (k < NBASIS) ? W0[k * HID + n] * 0.35355339f : 0.0f;  // 1/sqrt(8)
        sW0t[i] = f2bf(v);
    }
    for (int i = tid; i < HID * HID; i += 256) {
        int n = i >> 6, k = i & 63;
        sW1t[i] = f2bf(W1[k * HID + n] * 0.125f);    // 1/sqrt(64)
        sW2t[i] = f2bf(W2[k * HID + n] * 0.125f);
    }
    for (int i = tid; i < NIRR * HID; i += 256) {
        int n = i >> 6, k = i & 63;
        sW3t[i] = f2bf(W3[k * NIRR + n] * 0.03125f); // 1/sqrt(64) * 1/sqrt(16)
    }
    __syncthreads();

    const int wave = tid >> 5;
    const int lane = tid & 31;
    const int half = lane >> 4;      // 0: lanes 0-15, 1: lanes 16-31
    const int l15  = lane & 15;
    const int kb   = half * 8;       // lane-half K base inside a K=32 step
    const int edgeBase = blockIdx.x * EDGES_PER_BLOCK + wave * 16;
    unsigned short* act = &sAct[wave * 16 * HID];

    // ---- layer 1: radial [16 x 8 (pad 32)] @ W0t -> h [16 x 64] ----
    {
        Frag a;
#pragma unroll
        for (int i = 0; i < 16; ++i) a.s[i] = 0;
        if (half == 0) {
            const float* rp = radial + (size_t)(edgeBase + l15) * NBASIS;
#pragma unroll
            for (int i = 0; i < NBASIS; ++i) a.s[i] = f2bf(rp[i]);
        }
        v8f acc[4];
#pragma unroll
        for (int nb = 0; nb < 4; ++nb) {
            v8f c = {};
            acc[nb] = wmma_bf16(a, lds_frag(&sW0t[(nb * 16 + l15) * 32 + kb]), c);
        }
#pragma unroll
        for (int nb = 0; nb < 4; ++nb)
#pragma unroll
            for (int r = 0; r < 8; ++r)
                act[(half * 8 + r) * HID + nb * 16 + l15] = f2bf(silu_norm(acc[nb][r]));
        __syncthreads();
    }

    // ---- layers 2 & 3 ----
    hidden_layer(sW1t, act, l15, kb, half);
    hidden_layer(sW2t, act, l15, kb, half);

    // ---- layer 4: all 12 N-blocks (acc[0..3]=m0, [4..7]=m1, [8..11]=m2) ----
    v8f acc[12];
#pragma unroll
    for (int nb = 0; nb < 12; ++nb) acc[nb] = (v8f){};
    {
        Frag af[2];
#pragma unroll
        for (int ks = 0; ks < 2; ++ks)
            af[ks] = lds_frag(&act[l15 * HID + ks * 32 + kb]);
#pragma unroll
        for (int nb = 0; nb < 12; ++nb) {
#pragma unroll
            for (int ks = 0; ks < 2; ++ks)
                acc[nb] = wmma_bf16(af[ks],
                                    lds_frag(&sW3t[(nb * 16 + l15) * HID + ks * 32 + kb]),
                                    acc[nb]);
        }
    }

    // ---- epilogue: edges outer (geometry once per edge), feature blocks inner ----
#pragma unroll
    for (int r = 0; r < 8; ++r) {
        const int e   = edgeBase + half * 8 + r;   // edge row in C-layout
        const int snd = senders[e];
        const int rcv = receivers[e];

        float vx = vectors[e * 3 + 0];
        float vy = vectors[e * 3 + 1];
        float vz = vectors[e * 3 + 2];
        float ri = __builtin_amdgcn_rsqf(vx * vx + vy * vy + vz * vz);
        float ux = vx * ri, uy = vy * ri, uz = vz * ri;

        float y10 = SQRT3 * ux, y11 = SQRT3 * uy, y12 = SQRT3 * uz;
        float y20 = C15 * ux * uy;
        float y21 = C15 * uy * uz;
        float y22 = C5H * (3.0f * uz * uz - 1.0f);
        float y23 = C15 * ux * uz;
        float y24 = C15H * (ux * ux - uy * uy);

        const float* nfr = node_feats + (size_t)snd * MULF;
        float* ob = out + (size_t)rcv * OUTF;

#pragma unroll
        for (int nb0 = 0; nb0 < 4; ++nb0) {
            const int f = nb0 * 16 + l15;          // feature column this lane owns
            float nf = nfr[f];
            float m0 = acc[nb0 + 0][r] * nf;
            float m1 = acc[nb0 + 4][r] * nf;
            float m2 = acc[nb0 + 8][r] * nf;

            // chunk 0: msg0 * m0
            atomic_add_f32(ob + f, m0);
            // chunk 1: (msg0 (x) y1) * m1
            float* o1 = ob + MULF + f * 3;
            atomic_add_f32(o1 + 0, m1 * y10);
            atomic_add_f32(o1 + 1, m1 * y11);
            atomic_add_f32(o1 + 2, m1 * y12);
            // chunk 2: (msg0 (x) y2) * m2
            float* o2 = ob + (MULF + MULF * 3) + f * 5;
            atomic_add_f32(o2 + 0, m2 * y20);
            atomic_add_f32(o2 + 1, m2 * y21);
            atomic_add_f32(o2 + 2, m2 * y22);
            atomic_add_f32(o2 + 3, m2 * y23);
            atomic_add_f32(o2 + 4, m2 * y24);
        }
    }
}

extern "C" void kernel_launch(void* const* d_in, const int* in_sizes, int n_in,
                              void* d_out, int out_size, void* d_ws, size_t ws_size,
                              hipStream_t stream) {
    const float* vectors    = (const float*)d_in[0];
    const float* node_feats = (const float*)d_in[1];
    const float* radial     = (const float*)d_in[2];
    const float* W0         = (const float*)d_in[3];
    const float* W1         = (const float*)d_in[4];
    const float* W2         = (const float*)d_in[5];
    const float* W3         = (const float*)d_in[6];
    const int*   senders    = (const int*)d_in[7];
    const int*   receivers  = (const int*)d_in[8];
    float* out = (float*)d_out;

    const int E = in_sizes[0] / 3;   // vectors is (E,3)

    // output is accumulated with atomics -> must start from zero every call
    hipMemsetAsync(d_out, 0, (size_t)out_size * sizeof(float), stream);

    const int blocks = E / EDGES_PER_BLOCK;   // E = 262144 -> 2048 blocks
    mace_msgpass_kernel<<<dim3(blocks), dim3(256), 0, stream>>>(
        vectors, node_feats, radial, W0, W1, W2, W3, senders, receivers, out);
}
